// mimic_net_14345190768657
// MI455X (gfx1250) — matrix operand order repaired and checked by reading, not compiled, not run on hardware
//
#include <hip/hip_runtime.h>

// ---------------------------------------------------------------------------
// ROI-masked MSE mimic loss for MI455X (gfx1250).
//
// Roofline: dominant cost is streaming 2 * 4*1024*64*64 * 4B = 134 MB once
// (channel-summed squared diff). At 23.3 TB/s that's ~5.8 us; everything else
// touches <5 MB. Stage 1 therefore uses f32x4 non-temporal loads (read-once,
// no reuse -> don't pollute caches) + global_prefetch, with 2048 waves and
// 32 independent 16B loads in flight per thread.
//
// The ROI einsum sse[n] = my[n,:] . diff2 . mx[n,:]^T is done on the WMMA
// pipe: P(16x64) = My(16x64) x diff2(64x64) via v_wmma_f32_16x16x32_f16
// (masks are exact in f16; diff2 uses an f16 hi/lo split accumulated in f32
// for ~2^-22 relative error), then a per-row dot with mx + wave shuffle
// reduction. Operand VGPR layouts follow CDNA5 ISA 7.12.2 (16-bit A 16x32,
// B 32x16), gathered from LDS (diff2 stored transposed, padded stride 65).
// ---------------------------------------------------------------------------

typedef float     f32x4 __attribute__((ext_vector_type(4)));
typedef _Float16  v16h  __attribute__((ext_vector_type(16)));
typedef float     v8f   __attribute__((ext_vector_type(8)));

#define BATCH    4
#define CHANNELS 1024
#define HH       64
#define WW       64
#define HW       (HH * WW)          // 4096
#define N_ROIS   256
#define NCHUNK   64                 // channel chunks for stage-1 parallelism
#define CPC      (CHANNELS / NCHUNK) // 16 channels per chunk

// ---------------------------------------------------------------------------
// Stage 1: partial channel reduction of (std - tch)^2.
// grid = BATCH*NCHUNK = 256 blocks, 256 threads. Each block: one batch plane
// (4096 px as 1024 f32x4), 16 channels. Writes part[b][chunk][1024] f32x4.
// ---------------------------------------------------------------------------
__global__ __launch_bounds__(256) void diff2_partial_kernel(
    const float* __restrict__ s, const float* __restrict__ t,
    float* __restrict__ part)
{
    const int b   = blockIdx.x >> 6;      // / NCHUNK
    const int ch  = blockIdx.x & 63;      // % NCHUNK
    const int tid = threadIdx.x;

    const f32x4* sp = (const f32x4*)(s + (size_t)(b * CHANNELS + ch * CPC) * HW);
    const f32x4* tp = (const f32x4*)(t + (size_t)(b * CHANNELS + ch * CPC) * HW);

    f32x4 acc0 = {0.f, 0.f, 0.f, 0.f};
    f32x4 acc1 = acc0, acc2 = acc0, acc3 = acc0;

    #pragma unroll 4
    for (int c = 0; c < CPC; ++c) {
        const f32x4* sc = sp + (size_t)c * (HW / 4);
        const f32x4* tc = tp + (size_t)c * (HW / 4);
        // speculative prefetch two channel planes ahead -> global_prefetch_b8
        __builtin_prefetch((const void*)(sc + 2 * (HW / 4) + tid), 0, 0);
        __builtin_prefetch((const void*)(tc + 2 * (HW / 4) + tid), 0, 0);

        f32x4 d0 = __builtin_nontemporal_load(sc + tid      ) - __builtin_nontemporal_load(tc + tid      );
        f32x4 d1 = __builtin_nontemporal_load(sc + tid + 256) - __builtin_nontemporal_load(tc + tid + 256);
        f32x4 d2 = __builtin_nontemporal_load(sc + tid + 512) - __builtin_nontemporal_load(tc + tid + 512);
        f32x4 d3 = __builtin_nontemporal_load(sc + tid + 768) - __builtin_nontemporal_load(tc + tid + 768);
        acc0 += d0 * d0;
        acc1 += d1 * d1;
        acc2 += d2 * d2;
        acc3 += d3 * d3;
    }

    f32x4* pp = (f32x4*)part + (size_t)(b * NCHUNK + ch) * (HW / 4);
    pp[tid      ] = acc0;
    pp[tid + 256] = acc1;
    pp[tid + 512] = acc2;
    pp[tid + 768] = acc3;
}

// ---------------------------------------------------------------------------
// Stage 2: reduce the 64 chunk partials per pixel and pack diff2 into f16
// hi/lo planes (d = hi + lo, lo = f16(d - f32(hi))). grid = 4, 256 threads.
// ---------------------------------------------------------------------------
__global__ __launch_bounds__(256) void reduce_pack_kernel(
    const float* __restrict__ part,
    _Float16* __restrict__ hi, _Float16* __restrict__ lo)
{
    const int b   = blockIdx.x;
    const int tid = threadIdx.x;
    const f32x4* pp = (const f32x4*)part + (size_t)b * NCHUNK * (HW / 4);

    #pragma unroll
    for (int j = 0; j < 4; ++j) {
        const int p4 = tid + 256 * j;
        f32x4 sum = {0.f, 0.f, 0.f, 0.f};
        #pragma unroll 8
        for (int ch = 0; ch < NCHUNK; ++ch)
            sum += pp[(size_t)ch * (HW / 4) + p4];
        const int base = b * HW + p4 * 4;
        #pragma unroll
        for (int e = 0; e < 4; ++e) {
            const float d = sum[e];
            const _Float16 h = (_Float16)d;
            hi[base + e] = h;
            lo[base + e] = (_Float16)(d - (float)h);
        }
    }
}

// ---------------------------------------------------------------------------
// Stage 3: per-ROI masked sums via WMMA. One wave (32 threads) per block;
// grid = BATCH * (N_ROIS/16) = 64 blocks. EXEC is all-ones around the WMMAs.
//
//   P(16x64) = My(16x64) x diff2(64x64)     [4 N-tiles x 2 K-steps, hi+lo]
//   sse[n]   = sum_x P[n,x] * mx[n,x]       [VALU dot + 16-lane shuffle tree]
// ---------------------------------------------------------------------------
__global__ __launch_bounds__(32) void roi_wmma_kernel(
    const _Float16* __restrict__ hiP, const _Float16* __restrict__ loP,
    const float* __restrict__ rois, const int* __restrict__ pstride,
    float* __restrict__ roiBuf)
{
    __shared__ _Float16 dTh[64 * 65];   // diff2 hi, transposed [x][y], pad 65
    __shared__ _Float16 dTl[64 * 65];   // diff2 lo, transposed
    __shared__ _Float16 myL[16 * 64];   // row mask  my[n][y]
    __shared__ _Float16 mxL[16 * 64];   // col mask  mx[n][x]
    __shared__ int ix1[16], iy1[16], ix2[16], iy2[16];

    const int lane = threadIdx.x;
    const int b    = blockIdx.x >> 4;   // 16 ROI tiles per batch
    const int tile = blockIdx.x & 15;
    const int stride = *pstride;

    // --- ROI box -> feature-grid int coords (floor == trunc, coords >= 0) ---
    if (lane < 16) {
        const int n = b * N_ROIS + tile * 16 + lane;
        const float* r = rois + (size_t)n * 5;
        const float fs = (float)stride;
        ix1[lane] = (int)floorf(r[1] / fs);
        iy1[lane] = (int)floorf(r[2] / fs);
        ix2[lane] = (int)floorf(r[3] / fs);
        iy2[lane] = (int)floorf(r[4] / fs);
    }
    __syncthreads();

    // --- build separable 0/1 masks in f16 (exact) ---
    {
        const int i  = lane & 15;
        const int c0 = (lane >> 4) * 32;
        const int x1 = ix1[i], x2 = ix2[i], y1 = iy1[i], y2 = iy2[i];
        #pragma unroll 8
        for (int k = 0; k < 32; ++k) {
            const int c = c0 + k;
            myL[i * 64 + c] = (_Float16)((c >= y1 && c <= y2) ? 1.0f : 0.0f);
            mxL[i * 64 + c] = (_Float16)((c >= x1 && c <= x2) ? 1.0f : 0.0f);
        }
    }

    // --- stage diff2 hi/lo transposed into LDS (coalesced global reads) ---
    const _Float16* hp = hiP + (size_t)b * HW;
    const _Float16* lp = loP + (size_t)b * HW;
    for (int idx = lane; idx < HW; idx += 32) {
        const int y = idx >> 6, x = idx & 63;
        dTh[x * 65 + y] = hp[idx];
        dTl[x * 65 + y] = lp[idx];
    }
    __syncthreads();

    // --- gather A operand (mask My) per CDNA5 16-bit A 16x32 layout:
    //     lane l: M = l%16; VGPR v holds K = (v%4)*2 + (v/4)*16 + (l/16)*8 ---
    v16h a0, a1;
    const int M  = lane & 15;
    const int kh = (lane >> 4) * 8;
    #pragma unroll
    for (int v = 0; v < 8; ++v) {
        const int k0 = (v & 3) * 2 + (v >> 2) * 16 + kh;
        a0[2 * v    ] = myL[M * 64 + k0];
        a0[2 * v + 1] = myL[M * 64 + k0 + 1];
        a1[2 * v    ] = myL[M * 64 + 32 + k0];
        a1[2 * v + 1] = myL[M * 64 + 32 + k0 + 1];
    }

    // --- B operand (diff2) per 16-bit B 32x16 layout:
    //     lane l: N = l%16; VGPR v holds K = (v%4)*2 + (v/4)*16 + (l/16)*8 ---
    v8f acc[4];
    const int N  = lane & 15;
    const int kb = (lane >> 4) * 8;
    #pragma unroll
    for (int t = 0; t < 4; ++t) {
        const int x = t * 16 + N;
        v16h bh0, bh1, bl0, bl1;
        #pragma unroll
        for (int v = 0; v < 8; ++v) {
            const int k0 = (v & 3) * 2 + (v >> 2) * 16 + kb;
            bh0[2 * v    ] = dTh[x * 65 + k0];
            bh0[2 * v + 1] = dTh[x * 65 + k0 + 1];
            bh1[2 * v    ] = dTh[x * 65 + 32 + k0];
            bh1[2 * v + 1] = dTh[x * 65 + 32 + k0 + 1];
            bl0[2 * v    ] = dTl[x * 65 + k0];
            bl0[2 * v + 1] = dTl[x * 65 + k0 + 1];
            bl1[2 * v    ] = dTl[x * 65 + 32 + k0];
            bl1[2 * v + 1] = dTl[x * 65 + 32 + k0 + 1];
        }
        v8f c = {};
        c = __builtin_amdgcn_wmma_f32_16x16x32_f16(false, a0, false, bh0, (short)0, c, false, false);
        c = __builtin_amdgcn_wmma_f32_16x16x32_f16(false, a1, false, bh1, (short)0, c, false, false);
        c = __builtin_amdgcn_wmma_f32_16x16x32_f16(false, a0, false, bl0, (short)0, c, false, false);
        c = __builtin_amdgcn_wmma_f32_16x16x32_f16(false, a1, false, bl1, (short)0, c, false, false);
        acc[t] = c;
    }

    // --- finish: sse[n] = sum_x P[n,x]*mx[n,x]; C/D layout: row = r + 8*(l/16),
    //     col = l%16; reduce across each 16-lane half with xor shuffles ---
    #pragma unroll
    for (int r = 0; r < 8; ++r) {
        const int row = r + ((lane >> 4) << 3);   // local ROI index 0..15
        float val = 0.f;
        #pragma unroll
        for (int t = 0; t < 4; ++t) {
            const int x = t * 16 + N;
            val += acc[t][r] * (float)mxL[row * 64 + x];
        }
        for (int m = 1; m < 16; m <<= 1)
            val += __shfl_xor(val, m, 32);
        if ((lane & 15) == 0) {
            const int x1 = ix1[row], x2 = ix2[row], y1 = iy1[row], y2 = iy2[row];
            const bool valid = (y2 > y1) && (x2 > x1);
            const float area = (float)((y2 - y1 + 1) * (x2 - x1 + 1));
            const float out  = valid ? val / ((float)CHANNELS * area) : 0.0f;
            roiBuf[b * N_ROIS + tile * 16 + row] = out;
        }
    }
}

// ---------------------------------------------------------------------------
// Stage 4: mean over all B*N ROI values -> scalar loss.
// ---------------------------------------------------------------------------
__global__ __launch_bounds__(256) void final_reduce_kernel(
    const float* __restrict__ roiBuf, float* __restrict__ out)
{
    __shared__ float s[256];
    const int tid = threadIdx.x;
    s[tid] = roiBuf[tid] + roiBuf[tid + 256] + roiBuf[tid + 512] + roiBuf[tid + 768];
    __syncthreads();
    for (int w = 128; w > 0; w >>= 1) {
        if (tid < w) s[tid] += s[tid + w];
        __syncthreads();
    }
    if (tid == 0) out[0] = s[0] / (float)(BATCH * N_ROIS);
}

// ---------------------------------------------------------------------------
// kernel_launch: workspace layout (bytes):
//   [0, 4 MB)         part   : f32 [4][64][4096]
//   [+0, +32 KB)      hi     : f16 [4][4096]
//   [+32 KB, +64 KB)  lo     : f16 [4][4096]
//   [+64 KB, +68 KB)  roiBuf : f32 [1024]
// Total ~4.07 MB. Every ws byte read is written first (poison-safe).
// ---------------------------------------------------------------------------
extern "C" void kernel_launch(void* const* d_in, const int* in_sizes, int n_in,
                              void* d_out, int out_size, void* d_ws, size_t ws_size,
                              hipStream_t stream)
{
    const float* std_feat = (const float*)d_in[0];
    const float* tch_feat = (const float*)d_in[1];
    const float* rois     = (const float*)d_in[2];
    const int*   stride   = (const int*)d_in[3];
    float*       out      = (float*)d_out;

    char* ws = (char*)d_ws;
    float*    part   = (float*)ws;
    _Float16* hi     = (_Float16*)(ws + (size_t)BATCH * NCHUNK * HW * sizeof(float));
    _Float16* lo     = hi + (size_t)BATCH * HW;
    float*    roiBuf = (float*)(lo + (size_t)BATCH * HW);

    diff2_partial_kernel<<<BATCH * NCHUNK, 256, 0, stream>>>(std_feat, tch_feat, part);
    reduce_pack_kernel  <<<BATCH,          256, 0, stream>>>(part, hi, lo);
    roi_wmma_kernel     <<<BATCH * 16,      32, 0, stream>>>(hi, lo, rois, stride, roiBuf);
    final_reduce_kernel <<<1,              256, 0, stream>>>(roiBuf, out);
}
